// SpGAT_12730283066032
// MI455X (gfx1250) — matrix operand order repaired and checked by reading, not compiled, not run on hardware
//
#include <hip/hip_runtime.h>
#include <cstdint>
#include <cstddef>

// ---------------- problem constants ----------------
#define NN      50000      // nodes
#define NE      400000     // edges
#define NFEAT   128
#define NHID    64
#define RELD    128
#define NHEADS  4
#define NREL    500
#define D1      256        // NHEADS*NHID
#define ALPHA   0.2f

// ---------------- WMMA types ----------------
typedef __attribute__((ext_vector_type(16))) __bf16 v16bf;
typedef __attribute__((ext_vector_type(8)))  float  v8f;

union FragBF { v16bf bf; uint4 q[2]; };

__device__ __forceinline__ unsigned short f2bf(float f) {
  union { float f; unsigned int u; } x; x.f = f;
  unsigned int u = x.u;
  u = (u + 0x7FFFu + ((u >> 16) & 1u)) >> 16;   // RNE
  return (unsigned short)u;
}

// ---------------- utility kernels ----------------
__global__ __launch_bounds__(256) void zero_f32(float* p, size_t n) {
  size_t i = (size_t)blockIdx.x * blockDim.x + threadIdx.x;
  if (i < n) p[i] = 0.0f;
}

__global__ __launch_bounds__(256) void cvt_f32_bf16(const float* in, unsigned short* out, size_t n) {
  size_t i = (size_t)blockIdx.x * blockDim.x + threadIdx.x;
  if (i < n) out[i] = f2bf(in[i]);
}

// A1bf[c][k], c<512: col layout head*128 + half*64 + j   (half0=a_src, half1=a_dst)
__global__ __launch_bounds__(256) void pack_A1(const float* Ah, unsigned short* A1bf) {
  int idx = blockIdx.x * blockDim.x + threadIdx.x;      // 512*128
  if (idx >= 512 * 128) return;
  int c = idx >> 7, k = idx & 127;
  int head = c >> 7, half = (c >> 6) & 1, j = c & 63;
  A1bf[idx] = f2bf(Ah[(size_t)(head * 64 + j) * 384 + half * 128 + k]);
}

// ARbf[c][k], c<256: col = head*64 + j, weight = a_heads[head][j][256+k]
__global__ __launch_bounds__(256) void pack_AR(const float* Ah, unsigned short* ARbf) {
  int idx = blockIdx.x * blockDim.x + threadIdx.x;      // 256*128
  if (idx >= 256 * 128) return;
  int c = idx >> 7, k = idx & 127;
  int head = c >> 6, j = c & 63;
  ARbf[idx] = f2bf(Ah[(size_t)(head * 64 + j) * 384 + 256 + k]);
}

// A2bf[c][k], c<512: half=c>>8, j=c&255 ; weight = a_out[j][half*256 + k]
__global__ __launch_bounds__(256) void pack_A2(const float* Ao, unsigned short* A2bf) {
  int idx = blockIdx.x * blockDim.x + threadIdx.x;      // 512*256
  if (idx >= 512 * 256) return;
  int c = idx >> 8, k = idx & 255;
  int half = c >> 8, j = c & 255;
  A2bf[idx] = f2bf(Ao[(size_t)j * 768 + half * 256 + k]);
}

// ---------------- WMMA bf16 GEMM:  C[M,N] = A[M,K] * B[N,K]^T  (f32 accum) ----
// one wave computes a 16x64 output strip (1 M-tile x 4 N-tiles): the A fragment
// is loaded once per K-step and reused by 4 WMMAs.  M mult of 16, N mult of 64,
// K mult of 32.
__global__ __launch_bounds__(256) void wmma_gemm_bf16(const unsigned short* __restrict__ A,
                                                      const unsigned short* __restrict__ B,
                                                      float* __restrict__ C,
                                                      int M, int N, int K) {
  int wid  = (int)((blockIdx.x * blockDim.x + threadIdx.x) >> 5);
  int lane = threadIdx.x & 31;
  int ng   = N >> 6;                 // groups of 4 N-tiles
  int total = (M >> 4) * ng;
  if (wid >= total) return;
  int tm  = wid / ng;
  int tn0 = (wid % ng) * 4;          // first of 4 consecutive N-tiles

  int r16 = lane & 15;
  int hiA = (lane >> 4) * 16;        // A frag: lanes>=16 hold K=16..31 contiguous
  int hiB = (lane >> 4) * 16;        // B frag: lanes>=16 hold K=16..31 contiguous
  const unsigned short* ap = A + (size_t)(tm * 16 + r16) * K + hiA;
  const unsigned short* bp = B + (size_t)(tn0 * 16 + r16) * K + hiB;
  const size_t bstride = (size_t)16 * K;   // next N-tile's rows

  v8f acc0 = {}, acc1 = {}, acc2 = {}, acc3 = {};
  for (int k = 0; k < K; k += 32) {
    FragBF a, b0, b1, b2, b3;
    a.q[0]  = *(const uint4*)(ap + k);
    a.q[1]  = *(const uint4*)(ap + k + 8);
    b0.q[0] = *(const uint4*)(bp + k);
    b0.q[1] = *(const uint4*)(bp + k + 8);
    b1.q[0] = *(const uint4*)(bp + bstride + k);
    b1.q[1] = *(const uint4*)(bp + bstride + k + 8);
    b2.q[0] = *(const uint4*)(bp + 2 * bstride + k);
    b2.q[1] = *(const uint4*)(bp + 2 * bstride + k + 8);
    b3.q[0] = *(const uint4*)(bp + 3 * bstride + k);
    b3.q[1] = *(const uint4*)(bp + 3 * bstride + k + 8);
    acc0 = __builtin_amdgcn_wmma_f32_16x16x32_bf16(false, a.bf, false, b0.bf, (short)0, acc0, false, false);
    acc1 = __builtin_amdgcn_wmma_f32_16x16x32_bf16(false, a.bf, false, b1.bf, (short)0, acc1, false, false);
    acc2 = __builtin_amdgcn_wmma_f32_16x16x32_bf16(false, a.bf, false, b2.bf, (short)0, acc2, false, false);
    acc3 = __builtin_amdgcn_wmma_f32_16x16x32_bf16(false, a.bf, false, b3.bf, (short)0, acc3, false, false);
  }
  // C/D layout: VGPR r -> M = r (lanes 0-15) / r+8 (lanes 16-31), N = lane&15
  float* cp = C + (size_t)(tm * 16 + (lane >> 4) * 8) * N + tn0 * 16 + r16;
#pragma unroll
  for (int r = 0; r < 8; ++r) {
    cp[(size_t)r * N]      = acc0[r];
    cp[(size_t)r * N + 16] = acc1[r];
    cp[(size_t)r * N + 32] = acc2[r];
    cp[(size_t)r * N + 48] = acc3[r];
  }
}

// ---------------- layer-1 edge kernel: one wave per edge --------------------
// NM1: [NN,512] col = head*128 + half*64 + j ; R1: [NE,256] col = head*64 + j
__global__ __launch_bounds__(256) void edge_l1(const float* __restrict__ NM1,
                                               const float* __restrict__ R1,
                                               const int* __restrict__ src,
                                               const int* __restrict__ dst,
                                               const float* __restrict__ a2h, // [4*64]
                                               float* __restrict__ h1,        // [NN,256]
                                               float* __restrict__ rs1) {     // [NN,4]
  int e    = (int)((blockIdx.x * blockDim.x + threadIdx.x) >> 5);
  int lane = threadIdx.x & 31;
  if (e >= NE) return;
  int s = src[e], d = dst[e];
  const float* ms = NM1 + (size_t)s * 512;
  const float* md = NM1 + (size_t)d * 512;
  const float* rr = R1 + (size_t)e * 256;

  float em[8];
  float part[4] = {0.f, 0.f, 0.f, 0.f};
#pragma unroll
  for (int k = 0; k < 8; ++k) {
    int c = lane + 32 * k;           // head = k>>1 (uniform per k)
    int h = c >> 6, j = c & 63;
    float v = ms[h * 128 + j] + md[h * 128 + 64 + j] + rr[c];
    em[k] = v;
    part[k >> 1] += v * a2h[h * 64 + j];
  }
#pragma unroll
  for (int off = 16; off > 0; off >>= 1) {
    part[0] += __shfl_xor(part[0], off, 32);
    part[1] += __shfl_xor(part[1], off, 32);
    part[2] += __shfl_xor(part[2], off, 32);
    part[3] += __shfl_xor(part[3], off, 32);
  }
  float w[4];
#pragma unroll
  for (int h = 0; h < 4; ++h) {
    float p  = part[h];
    float lr = p > 0.f ? p : ALPHA * p;
    w[h] = __expf(-lr);
  }
#pragma unroll
  for (int k = 0; k < 8; ++k) {
    int c = lane + 32 * k;
    unsafeAtomicAdd(&h1[(size_t)s * 256 + c], w[c >> 6] * em[k]);
  }
  if (lane < 4) unsafeAtomicAdd(&rs1[(size_t)s * 4 + lane], w[lane]);
}

// normalize + ELU -> bf16 x  (feeds layer-2 GEMM)
__global__ __launch_bounds__(256) void fin_l1(const float* __restrict__ h1,
                                              const float* __restrict__ rs1,
                                              unsigned short* __restrict__ xbf) {
  size_t i = (size_t)blockIdx.x * blockDim.x + threadIdx.x;
  if (i >= (size_t)NN * 256) return;
  size_t n = i >> 8;
  int c = (int)(i & 255);
  float v = h1[i] / (rs1[n * 4 + (c >> 6)] + 1e-16f);
  v = v > 0.f ? v : (__expf(v) - 1.f);
  xbf[i] = f2bf(v);
}

// ---------------- small relation GEMMs (500 rows, scalar) -------------------
__global__ __launch_bounds__(256) void outrel_k(const float* __restrict__ RelE,
                                                const float* __restrict__ W1,
                                                float* __restrict__ outrel) {
  int idx = blockIdx.x * blockDim.x + threadIdx.x;     // NREL*256
  if (idx >= NREL * 256) return;
  int r = idx >> 8, c = idx & 255;
  float sAcc = 0.f;
  for (int k = 0; k < RELD; ++k) sAcc += RelE[r * RELD + k] * W1[k * 256 + c];
  outrel[idx] = sAcc;
}

__global__ __launch_bounds__(256) void rel2_k(const float* __restrict__ outrel,
                                              const float* __restrict__ Ao,
                                              float* __restrict__ rel2) {
  int idx = blockIdx.x * blockDim.x + threadIdx.x;     // NREL*256
  if (idx >= NREL * 256) return;
  int r = idx >> 8, c = idx & 255;
  float sAcc = 0.f;
  for (int j = 0; j < 256; ++j) sAcc += outrel[r * 256 + j] * Ao[(size_t)c * 768 + 512 + j];
  rel2[idx] = sAcc;
}

// ---------------- layer-2 edge kernel: one wave per edge --------------------
// M2: [NN,512] col = half*256 + c
__global__ __launch_bounds__(256) void edge_l2(const float* __restrict__ M2,
                                               const float* __restrict__ rel2,
                                               const int* __restrict__ src,
                                               const int* __restrict__ dst,
                                               const int* __restrict__ etype,
                                               const float* __restrict__ a2o, // [256]
                                               float* __restrict__ h2,        // [NN,256]
                                               float* __restrict__ rs2) {     // [NN]
  int e    = (int)((blockIdx.x * blockDim.x + threadIdx.x) >> 5);
  int lane = threadIdx.x & 31;
  if (e >= NE) return;
  int s = src[e], d = dst[e], t = etype[e];
  const float* ms = M2 + (size_t)s * 512;
  const float* md = M2 + (size_t)d * 512 + 256;
  const float* rr = rel2 + (size_t)t * 256;

  float em[8];
  float part = 0.f;
#pragma unroll
  for (int k = 0; k < 8; ++k) {
    int c = lane + 32 * k;
    float v = ms[c] + md[c] + rr[c];
    em[k] = v;
    part += v * a2o[c];
  }
#pragma unroll
  for (int off = 16; off > 0; off >>= 1) part += __shfl_xor(part, off, 32);
  float lr = part > 0.f ? part : ALPHA * part;
  float w  = __expf(-lr);
#pragma unroll
  for (int k = 0; k < 8; ++k)
    unsafeAtomicAdd(&h2[(size_t)s * 256 + lane + 32 * k], w * em[k]);
  if (lane == 0) unsafeAtomicAdd(&rs2[s], w);
}

__global__ __launch_bounds__(256) void fin_l2(const float* __restrict__ h2,
                                              const float* __restrict__ rs2,
                                              float* __restrict__ out) {
  size_t i = (size_t)blockIdx.x * blockDim.x + threadIdx.x;
  if (i >= (size_t)NN * 256) return;
  float v = h2[i] / (rs2[i >> 8] + 1e-16f);
  out[i] = v > 0.f ? v : (__expf(v) - 1.f);
}

// ---------------- host launch ----------------
extern "C" void kernel_launch(void* const* d_in, const int* in_sizes, int n_in,
                              void* d_out, int out_size, void* d_ws, size_t ws_size,
                              hipStream_t stream) {
  (void)in_sizes; (void)n_in; (void)out_size; (void)ws_size;
  const float* H    = (const float*)d_in[0];   // [NN,128]
  const float* RelE = (const float*)d_in[1];   // [NREL,128]
  const float* EE   = (const float*)d_in[2];   // [NE,128]
  const float* Ah   = (const float*)d_in[3];   // [4,64,384]
  const float* A2h  = (const float*)d_in[4];   // [4,1,64]
  const float* Ao   = (const float*)d_in[5];   // [256,768]
  const float* A2o  = (const float*)d_in[6];   // [1,256]
  const float* W1   = (const float*)d_in[7];   // [128,256]
  const int* elist  = (const int*)d_in[8];     // [2,NE]
  const int* etype  = (const int*)d_in[9];     // [NE]
  const int* src = elist;
  const int* dst = elist + NE;

  char* ws = (char*)d_ws;
  size_t off = 0;
  auto alloc = [&](size_t bytes) -> char* {
    off = (off + 255) & ~(size_t)255;
    char* p = ws + off;
    off += bytes;
    return p;
  };
  unsigned short* Hbf  = (unsigned short*)alloc((size_t)NN * 128 * 2);
  unsigned short* Ebf  = (unsigned short*)alloc((size_t)NE * 128 * 2);
  unsigned short* A1bf = (unsigned short*)alloc((size_t)512 * 128 * 2);
  unsigned short* ARbf = (unsigned short*)alloc((size_t)256 * 128 * 2);
  unsigned short* A2bf = (unsigned short*)alloc((size_t)512 * 256 * 2);
  float* NM1 = (float*)alloc((size_t)NN * 512 * 4);   // later reused as M2
  float* R1  = (float*)alloc((size_t)NE * 256 * 4);   // later reused as h2+rs2
  float* h1  = (float*)alloc((size_t)NN * 256 * 4);
  float* rs1 = (float*)alloc((size_t)NN * 4 * 4);     // contiguous after h1
  unsigned short* xbf = (unsigned short*)alloc((size_t)NN * 256 * 2);
  float* outrel = (float*)alloc((size_t)NREL * 256 * 4);
  float* rel2   = (float*)alloc((size_t)NREL * 256 * 4);
  float* M2  = NM1;                                   // alias (NM1 dead by then)
  float* h2  = R1;                                    // alias (R1 dead by then)
  float* rs2 = R1 + (size_t)NN * 256;

  const int T = 256;
  // 1) bf16 conversions + weight packs
  cvt_f32_bf16<<<(unsigned)(((size_t)NN * 128 + T - 1) / T), T, 0, stream>>>(H, Hbf, (size_t)NN * 128);
  cvt_f32_bf16<<<(unsigned)(((size_t)NE * 128 + T - 1) / T), T, 0, stream>>>(EE, Ebf, (size_t)NE * 128);
  pack_A1<<<(512 * 128 + T - 1) / T, T, 0, stream>>>(Ah, A1bf);
  pack_AR<<<(256 * 128 + T - 1) / T, T, 0, stream>>>(Ah, ARbf);
  pack_A2<<<(512 * 256 + T - 1) / T, T, 0, stream>>>(Ao, A2bf);

  // 2) zero accumulators (h1 and rs1 are contiguous)
  zero_f32<<<(unsigned)(((size_t)NN * 260 + T - 1) / T), T, 0, stream>>>(h1, (size_t)NN * 260);

  // 3) WMMA GEMMs (each wave: 16x64 strip = 1 M-tile x 4 N-tiles)
  {
    int waves = (NN / 16) * (512 / 64);   // 25000
    wmma_gemm_bf16<<<(waves * 32 + T - 1) / T, T, 0, stream>>>(Hbf, A1bf, NM1, NN, 512, 128);
  }
  {
    int waves = (NE / 16) * (256 / 64);   // 100000
    wmma_gemm_bf16<<<(waves * 32 + T - 1) / T, T, 0, stream>>>(Ebf, ARbf, R1, NE, 256, 128);
  }

  // 4) layer-1 edge attention (one wave per edge)
  edge_l1<<<(NE * 32 + T - 1) / T, T, 0, stream>>>(NM1, R1, src, dst, A2h, h1, rs1);

  // 5) normalize + ELU -> bf16 x
  fin_l1<<<(unsigned)(((size_t)NN * 256 + T - 1) / T), T, 0, stream>>>(h1, rs1, xbf);

  // 6) M2 = x*A2^T [50000x512x256]  (aliases NM1)
  {
    int waves = (NN / 16) * (512 / 64);   // 25000
    wmma_gemm_bf16<<<(waves * 32 + T - 1) / T, T, 0, stream>>>(xbf, A2bf, M2, NN, 512, 256);
  }

  // 7) relation projections (tiny)
  outrel_k<<<(NREL * 256 + T - 1) / T, T, 0, stream>>>(RelE, W1, outrel);
  rel2_k<<<(NREL * 256 + T - 1) / T, T, 0, stream>>>(outrel, Ao, rel2);

  // 8) zero layer-2 accumulators (h2, rs2 contiguous)
  zero_f32<<<(unsigned)(((size_t)NN * 257 + T - 1) / T), T, 0, stream>>>(h2, (size_t)NN * 257);

  // 9) layer-2 edge attention
  edge_l2<<<(NE * 32 + T - 1) / T, T, 0, stream>>>(M2, rel2, src, dst, etype, A2o, h2, rs2);

  // 10) final normalize + ELU
  fin_l2<<<(unsigned)(((size_t)NN * 256 + T - 1) / T), T, 0, stream>>>(h2, rs2, (float*)d_out);
}
